// multihead_attention_89988154786028
// MI455X (gfx1250) — matrix operand order repaired and checked by reading, not compiled, not run on hardware
//
#include <hip/hip_runtime.h>
#include <hip/hip_bf16.h>

// Problem constants (match reference)
#define B_  4
#define S_  2048
#define U_  1024
#define H_  16
#define DH_ 64

typedef __bf16 bf16;
typedef __attribute__((ext_vector_type(16))) __bf16 v16bf;
typedef __attribute__((ext_vector_type(8)))  float  v8f;

// Direct global->LDS async copy path (CDNA5): bypasses VGPRs, tracked by ASYNCcnt.
#if defined(__gfx1250__) && __has_builtin(__builtin_amdgcn_global_load_async_to_lds_b128)
#define HAVE_ASYNC_LDS 1
// Builtin signature (from hipcc diagnostic): params are pointers to 16-byte int
// vectors: (v4i addrspace(1)*, v4i addrspace(3)*, imm offset, imm cpol).
typedef int v4i_b128 __attribute__((vector_size(4 * sizeof(int))));
#endif

__device__ __forceinline__ bf16 f2bf(float f) {
    union { float f; unsigned u; } v; v.f = f;
    unsigned r = v.u + 0x7FFFu + ((v.u >> 16) & 1u);   // round-to-nearest-even
    unsigned short h = (unsigned short)(r >> 16);
    return __builtin_bit_cast(bf16, h);
}

// ---------------------------------------------------------------------------
// Kernel 1: C = relu(X @ W + bias). X:[M,K] f32, W:[K,N] f32.
// Writes bf16 (for attention operands) and/or f32 (for residual branch).
// Block = 256 threads (8 waves). Block tile 128x128, K-step 32.
// Per wave: 16 rows x 128 cols = 8 accumulators of v8f.
// ---------------------------------------------------------------------------
__global__ __launch_bounds__(256)
void gemm_bias_relu_k(const float* __restrict__ X, const float* __restrict__ W,
                      const float* __restrict__ bias,
                      bf16* __restrict__ outb, float* __restrict__ outf,
                      int M, int N, int K)
{
    __shared__ __align__(16) bf16 Asm [128 * 32];   // [row][k]  row-major
    __shared__ __align__(16) bf16 BsmT[128 * 32];   // [n][k]    transposed W tile

    const int t    = threadIdx.x;
    const int lane = t & 31;
    const int wave = t >> 5;
    const bool hi  = lane >= 16;
    const int lm   = lane & 15;
    const int m0   = blockIdx.y * 128;
    const int n0   = blockIdx.x * 128;

    v8f acc[8];
#pragma unroll
    for (int n = 0; n < 8; ++n)
#pragma unroll
        for (int j = 0; j < 8; ++j) acc[n][j] = 0.0f;

    const int ksteps = K / 32;
    for (int kt = 0; kt < ksteps; ++kt) {
        __syncthreads();
        // ---- stage A tile (128x32), f32 -> bf16, row-major ----
#pragma unroll
        for (int i = 0; i < 4; ++i) {
            int idx = t + i * 256;            // 0..1023 float4 slots
            int row = idx >> 3, c4 = idx & 7; // 8 float4 per row
            float4 f = ((const float4*)(X + (size_t)(m0 + row) * K + kt * 32))[c4];
            bf16* d = &Asm[row * 32 + c4 * 4];
            d[0] = f2bf(f.x); d[1] = f2bf(f.y); d[2] = f2bf(f.z); d[3] = f2bf(f.w);
        }
        // ---- stage B tile (32x128) transposed into [n][k] ----
#pragma unroll
        for (int i = 0; i < 4; ++i) {
            int idx = t + i * 256;
            int k = idx >> 5, c4 = idx & 31;  // 32 float4 per k-row
            float4 f = ((const float4*)(W + (size_t)(kt * 32 + k) * N + n0))[c4];
            int n = c4 * 4;
            BsmT[(n + 0) * 32 + k] = f2bf(f.x);
            BsmT[(n + 1) * 32 + k] = f2bf(f.y);
            BsmT[(n + 2) * 32 + k] = f2bf(f.z);
            BsmT[(n + 3) * 32 + k] = f2bf(f.w);
        }
        if (kt + 1 < ksteps) {
            __builtin_prefetch(X + (size_t)(m0 + (t >> 1)) * K + (kt + 1) * 32, 0, 1);
            __builtin_prefetch(W + (size_t)((kt + 1) * 32 + (t & 31)) * N + n0, 0, 1);
        }
        __syncthreads();

        // ---- A fragment (16x32 bf16, ISA 7.12.2 layout) ----
        v16bf a;
        {
            const bf16* p = &Asm[(wave * 16 + lm) * 32 + (hi ? 8 : 0)];
#pragma unroll
            for (int i = 0; i < 8; ++i) { a[i] = p[i]; a[8 + i] = p[16 + i]; }
        }
        // ---- 8 WMMAs across N ----
#pragma unroll
        for (int nt = 0; nt < 8; ++nt) {
            v16bf bfr;
            const bf16* p = &BsmT[(nt * 16 + lm) * 32 + (hi ? 8 : 0)];
#pragma unroll
            for (int i = 0; i < 16; ++i) bfr[i] = p[i < 8 ? i : i + 8];
            acc[nt] = __builtin_amdgcn_wmma_f32_16x16x32_bf16(
                false, a, false, bfr, (short)0, acc[nt], false, false);
        }
    }

    // ---- epilogue: bias + relu, write bf16 and/or f32 ----
#pragma unroll
    for (int nt = 0; nt < 8; ++nt) {
#pragma unroll
        for (int v = 0; v < 8; ++v) {
            int row = m0 + wave * 16 + v + (hi ? 8 : 0);
            int col = n0 + nt * 16 + lm;
            float val = fmaxf(acc[nt][v] + bias[col], 0.0f);
            size_t off = (size_t)row * N + col;
            if (outb) outb[off] = f2bf(val);
            if (outf) outf[off] = val;
        }
    }
}

// ---------------------------------------------------------------------------
// Kernel 2: streaming (flash) attention per (b, h).
// Block = 128 threads (4 waves). Each block: 64 query rows; each wave: 16 rows.
// Loops over 64-key blocks with online softmax; O accumulated in f32.
// K tile staged via async global->LDS (ASYNCcnt) when the toolchain exposes it.
// ---------------------------------------------------------------------------
__global__ __launch_bounds__(128)
void flash_attn_k(const bf16* __restrict__ Qp, const bf16* __restrict__ Kp,
                  const bf16* __restrict__ Vp, float* __restrict__ Op)
{
    __shared__ __align__(16) bf16 Ksm [64 * 64];     // [key][dh]   (== B^T layout for QK^T)
    __shared__ __align__(16) bf16 VsmT[64 * 64];     // [dh][key]   (B-layout friendly for PV)
    __shared__ __align__(16) bf16 Psm [4 * 16 * 64]; // per-wave P tile, row-major

    const int t    = threadIdx.x;
    const int lane = t & 31;
    const int wave = t >> 5;
    const bool hi  = lane >= 16;
    const int lm   = lane & 15;
    const int q0   = blockIdx.x * 64;
    const int b    = blockIdx.y / H_;
    const int h    = blockIdx.y % H_;
    const float scale = 0.125f;  // 1/sqrt(DH)

    // ---- Q fragments, loaded once straight from global ----
    const int qrow = q0 + wave * 16 + lm;
    const bf16* qbase = Qp + ((size_t)(b * S_ + qrow) * U_ + h * DH_);
    v16bf qa[2];
#pragma unroll
    for (int ks = 0; ks < 2; ++ks) {
        const bf16* p = qbase + ks * 32 + (hi ? 8 : 0);
#pragma unroll
        for (int i = 0; i < 8; ++i) { qa[ks][i] = p[i]; qa[ks][8 + i] = p[16 + i]; }
    }

    float m_r[8], l_r[8];
    v8f o[4];
#pragma unroll
    for (int v = 0; v < 8; ++v) { m_r[v] = -3.402823e38f; l_r[v] = 0.0f; }
#pragma unroll
    for (int nt = 0; nt < 4; ++nt)
#pragma unroll
        for (int j = 0; j < 8; ++j) o[nt][j] = 0.0f;

    for (int kb = 0; kb < S_ / 64; ++kb) {
        __syncthreads();
        // ---- stage K block 64x64 (row-major, 16B granules) ----
#ifdef HAVE_ASYNC_LDS
#pragma unroll
        for (int i = 0; i < 4; ++i) {
            int idx = t + i * 128;              // 0..511 uint4 slots, 8 per key row
            int key = idx >> 3, c = idx & 7;
            const bf16* gsrc =
                Kp + ((size_t)(b * S_ + kb * 64 + key) * U_ + h * DH_) + c * 8;
            __builtin_amdgcn_global_load_async_to_lds_b128(
                (__attribute__((address_space(1))) v4i_b128*)(void*)gsrc,
                (__attribute__((address_space(3))) v4i_b128*)(void*)&Ksm[idx * 8],
                0, 0);
        }
#else
#pragma unroll
        for (int i = 0; i < 4; ++i) {
            int idx = t + i * 128;
            int key = idx >> 3, c = idx & 7;
            const uint4* src =
                (const uint4*)(Kp + ((size_t)(b * S_ + kb * 64 + key) * U_ + h * DH_));
            ((uint4*)Ksm)[idx] = src[c];
        }
#endif
        // ---- stage V block transposed: VsmT[dh][key] ----
#pragma unroll
        for (int i = 0; i < 32; ++i) {
            int idx = i * 128 + t;              // 0..4095
            int key = idx >> 6, dh = idx & 63;
            VsmT[dh * 64 + key] =
                Vp[(size_t)(b * S_ + kb * 64 + key) * U_ + h * DH_ + dh];
        }
#ifdef HAVE_ASYNC_LDS
#if __has_builtin(__builtin_amdgcn_s_wait_asynccnt)
        __builtin_amdgcn_s_wait_asynccnt(0);
#else
        asm volatile("s_wait_asynccnt 0x0" ::: "memory");
#endif
#endif
        __syncthreads();

        // ---- S = Q @ K^T  (16x64 per wave) ----
        v8f sacc[4];
#pragma unroll
        for (int nt = 0; nt < 4; ++nt)
#pragma unroll
            for (int j = 0; j < 8; ++j) sacc[nt][j] = 0.0f;
#pragma unroll
        for (int ks = 0; ks < 2; ++ks) {
#pragma unroll
            for (int nt = 0; nt < 4; ++nt) {
                v16bf kb_;
                const bf16* p = &Ksm[(nt * 16 + lm) * 64 + ks * 32 + (hi ? 8 : 0)];
#pragma unroll
                for (int i = 0; i < 16; ++i) kb_[i] = p[i < 8 ? i : i + 8];
                sacc[nt] = __builtin_amdgcn_wmma_f32_16x16x32_bf16(
                    false, qa[ks], false, kb_, (short)0, sacc[nt], false, false);
            }
        }

        // ---- online softmax (row-wise: per-lane rows + 16-lane reductions) ----
        float pvals[4][8];
#pragma unroll
        for (int v = 0; v < 8; ++v) {
            float rowm = -3.402823e38f;
#pragma unroll
            for (int nt = 0; nt < 4; ++nt) rowm = fmaxf(rowm, sacc[nt][v] * scale);
#pragma unroll
            for (int off = 1; off < 16; off <<= 1)
                rowm = fmaxf(rowm, __shfl_xor(rowm, off, 32));
            float m_new = fmaxf(m_r[v], rowm);
            float alpha = __expf(m_r[v] - m_new);
            float rs = 0.0f;
#pragma unroll
            for (int nt = 0; nt < 4; ++nt) {
                float pv = __expf(sacc[nt][v] * scale - m_new);
                pvals[nt][v] = pv;
                rs += pv;
            }
#pragma unroll
            for (int off = 1; off < 16; off <<= 1) rs += __shfl_xor(rs, off, 32);
            l_r[v] = l_r[v] * alpha + rs;
            m_r[v] = m_new;
#pragma unroll
            for (int nt = 0; nt < 4; ++nt) o[nt][v] *= alpha;
        }

        // ---- P: C-layout -> row-major LDS -> A-layout fragments ----
#pragma unroll
        for (int nt = 0; nt < 4; ++nt)
#pragma unroll
            for (int v = 0; v < 8; ++v)
                Psm[wave * 1024 + (v + (hi ? 8 : 0)) * 64 + nt * 16 + lm] =
                    f2bf(pvals[nt][v]);
        __syncthreads();

        // ---- O += P @ V ----
#pragma unroll
        for (int ks2 = 0; ks2 < 2; ++ks2) {
            v16bf pa;
            const bf16* pp = &Psm[wave * 1024 + lm * 64 + ks2 * 32 + (hi ? 8 : 0)];
#pragma unroll
            for (int i = 0; i < 8; ++i) { pa[i] = pp[i]; pa[8 + i] = pp[16 + i]; }
#pragma unroll
            for (int nt = 0; nt < 4; ++nt) {
                v16bf vb;
                const bf16* p = &VsmT[(nt * 16 + lm) * 64 + ks2 * 32 + (hi ? 8 : 0)];
#pragma unroll
                for (int i = 0; i < 16; ++i) vb[i] = p[i < 8 ? i : i + 8];
                o[nt] = __builtin_amdgcn_wmma_f32_16x16x32_bf16(
                    false, pa, false, vb, (short)0, o[nt], false, false);
            }
        }
    }

    // ---- normalize and write O (f32) ----
#pragma unroll
    for (int v = 0; v < 8; ++v) {
        float inv = 1.0f / l_r[v];
        int row = q0 + wave * 16 + v + (hi ? 8 : 0);
#pragma unroll
        for (int nt = 0; nt < 4; ++nt) {
            int col = h * DH_ + nt * 16 + lm;
            Op[(size_t)(b * S_ + row) * U_ + col] = o[nt][v] * inv;
        }
    }
}

// ---------------------------------------------------------------------------
// Kernel 3: out = gamma * normalize(relu(attn + vres)) + beta, one row/block.
// ---------------------------------------------------------------------------
__global__ __launch_bounds__(256)
void add_relu_ln_k(const float* __restrict__ attn, const float* __restrict__ vres,
                   const float* __restrict__ gamma, const float* __restrict__ beta,
                   float* __restrict__ out)
{
    __shared__ float redS[8], redQ[8];
    const int t = threadIdx.x;
    const size_t base = (size_t)blockIdx.x * U_;

    float x[4], s = 0.0f, q = 0.0f;
#pragma unroll
    for (int j = 0; j < 4; ++j) {
        int c = t + j * 256;
        float v = fmaxf(attn[base + c] + vres[base + c], 0.0f);
        x[j] = v; s += v; q += v * v;
    }
#pragma unroll
    for (int off = 16; off >= 1; off >>= 1) {
        s += __shfl_xor(s, off, 32);
        q += __shfl_xor(q, off, 32);
    }
    const int lane = t & 31, wave = t >> 5;
    if (lane == 0) { redS[wave] = s; redQ[wave] = q; }
    __syncthreads();
    float ts = 0.0f, tq = 0.0f;
#pragma unroll
    for (int w = 0; w < 8; ++w) { ts += redS[w]; tq += redQ[w]; }
    const float mean = ts * (1.0f / U_);
    const float var  = tq * (1.0f / U_) - mean * mean;
    const float rstd = rsqrtf(var + 1e-8f);
#pragma unroll
    for (int j = 0; j < 4; ++j) {
        int c = t + j * 256;
        out[base + c] = gamma[c] * ((x[j] - mean) * rstd) + beta[c];
    }
}

// ---------------------------------------------------------------------------
extern "C" void kernel_launch(void* const* d_in, const int* in_sizes, int n_in,
                              void* d_out, int out_size, void* d_ws, size_t ws_size,
                              hipStream_t stream)
{
    const float* queries = (const float*)d_in[0];
    const float* keys    = (const float*)d_in[1];
    const float* values  = (const float*)d_in[2];
    const float* Wq = (const float*)d_in[3];
    const float* bq = (const float*)d_in[4];
    const float* Wk = (const float*)d_in[5];
    const float* bk = (const float*)d_in[6];
    const float* Wv = (const float*)d_in[7];
    const float* bv = (const float*)d_in[8];
    const float* Wr = (const float*)d_in[9];
    const float* br = (const float*)d_in[10];
    const float* gamma = (const float*)d_in[11];
    const float* beta  = (const float*)d_in[12];
    float* out = (float*)d_out;

    char* ws = (char*)d_ws;
    const size_t actBf = (size_t)B_ * S_ * U_ * 2;               // 16 MB
    const size_t actF  = (size_t)B_ * S_ * U_ * sizeof(float);   // 32 MB
    bf16*  Qbf  = (bf16*)(ws);
    bf16*  Kbf  = (bf16*)(ws + actBf);
    bf16*  Vbf  = (bf16*)(ws + 2 * actBf);
    float* Vres = (float*)(ws + 3 * actBf);
    float* attn = (float*)(ws + 3 * actBf + actF);

    const int M = B_ * S_;
    dim3 ggrid(U_ / 128, M / 128);   // (8, 64)
    gemm_bias_relu_k<<<ggrid, 256, 0, stream>>>(queries, Wq, bq, Qbf, nullptr, M, U_, U_);
    gemm_bias_relu_k<<<ggrid, 256, 0, stream>>>(keys,    Wk, bk, Kbf, nullptr, M, U_, U_);
    gemm_bias_relu_k<<<ggrid, 256, 0, stream>>>(values,  Wv, bv, Vbf, nullptr, M, U_, U_);
    gemm_bias_relu_k<<<ggrid, 256, 0, stream>>>(values,  Wr, br, nullptr, Vres, M, U_, U_);

    flash_attn_k<<<dim3(S_ / 64, B_ * H_), 128, 0, stream>>>(Qbf, Kbf, Vbf, attn);

    add_relu_ln_k<<<M, 256, 0, stream>>>(attn, Vres, gamma, beta, out);
}
